// GNNStack_14078902796481
// MI455X (gfx1250) — compile-verified, matrix operand order checked by reading, not run on hardware
//
#include <hip/hip_runtime.h>
#include <hip/hip_bf16.h>

#define HEADS 2
#define HID   64
#define FDIM  128      // HEADS*HID
#define OUTC  40
#define SLOPE 0.2f
#define EPSV  1e-16f

typedef __bf16 bf16_t;
typedef bf16_t v16bf __attribute__((ext_vector_type(16)));
typedef bf16_t v8bf  __attribute__((ext_vector_type(8)));
typedef float  v8f   __attribute__((ext_vector_type(8)));

// ---------------------------------------------------------------- utilities
__global__ void k_cvt_f32_bf16(const float* __restrict__ src,
                               bf16_t* __restrict__ dst, int n, int relu) {
  int i = blockIdx.x * blockDim.x + threadIdx.x;
  if (i < n) {
    float v = src[i];
    if (relu) v = v > 0.f ? v : 0.f;
    dst[i] = (bf16_t)v;
  }
}

// convert W[Kout,K] fp32 -> Wb[KoutPad,K] bf16, zero rows >= Kout
__global__ void k_cvt_pad_bf16(const float* __restrict__ src,
                               bf16_t* __restrict__ dst,
                               int Kout, int K, int KoutPad) {
  int i = blockIdx.x * blockDim.x + threadIdx.x;
  if (i >= KoutPad * K) return;
  int r = i / K;
  dst[i] = (bf16_t)((r < Kout) ? src[i] : 0.0f);
}

__global__ void k_fill_f32(float* __restrict__ p, float v, int n) {
  int i = blockIdx.x * blockDim.x + threadIdx.x;
  if (i < n) p[i] = v;
}

// monotone float <-> uint mapping so atomicMax(u32) orders floats correctly
__device__ __forceinline__ unsigned ford(float f) {
  unsigned u = __float_as_uint(f);
  return (u & 0x80000000u) ? ~u : (u | 0x80000000u);
}
__device__ __forceinline__ float funord(unsigned u) {
  return (u & 0x80000000u) ? __uint_as_float(u & 0x7fffffffu)
                           : __uint_as_float(~u);
}

// ---------------------------------------------------------- WMMA bf16 GEMM
// C[N,KOUT] = A[N,K](bf16) x Wb[KOUTPAD,K](bf16)^T + bias, fp32 out.
// Wb pre-padded to KOUTPAD rows so all fragment loads are unguarded.
// Block: 256 threads = 8 waves; block covers 128 rows; each wave owns a
// 16-row strip and sweeps 16-col output tiles. A fragments are loaded once
// and reused across tiles; per tile all B fragments are loaded before the
// WMMA chain so ds-load latency overlaps the matrix pipe. The uniform
// N-bounds branch is hoisted outside the tile loop.
template <int K, int KOUTPAD, int KOUT>
__launch_bounds__(256)
__global__ void k_gemm_wmma(const bf16_t* __restrict__ A,
                            const bf16_t* __restrict__ Wb,
                            const float* __restrict__ bias,
                            float* __restrict__ C,
                            int N) {
  constexpr int LSTR  = K + 8;          // row stride (keeps 16B alignment)
  constexpr int KSTEP = K / 32;         // WMMA k-steps
  __shared__ bf16_t Xs[128][LSTR];
  __shared__ bf16_t Ws[KOUTPAD][LSTR];

  const int tid = threadIdx.x;
  const int rowBase = blockIdx.x * 128;

  // stage X tile (128 x K), zero-fill OOB rows
  constexpr int NCX = (128 * K) / 8;
#pragma unroll 2
  for (int ci = tid; ci < NCX; ci += 256) {
    int row = (ci * 8) / K;
    int col = (ci * 8) % K;
    int g = rowBase + row;
    v8bf v;
    if (g < N) {
      v = *(const v8bf*)(A + (size_t)g * K + col);
    } else {
#pragma unroll
      for (int j = 0; j < 8; ++j) v[j] = (bf16_t)0.0f;
    }
    *(v8bf*)&Xs[row][col] = v;
  }
  // stage W tile (KOUTPAD x K) — pre-padded, always in-bounds
  constexpr int NCW = (KOUTPAD * K) / 8;
#pragma unroll 2
  for (int ci = tid; ci < NCW; ci += 256) {
    int row = (ci * 8) / K;
    int col = (ci * 8) % K;
    *(v8bf*)&Ws[row][col] = *(const v8bf*)(Wb + (size_t)row * K + col);
  }
  __syncthreads();

  const int wave   = tid >> 5;
  const int lane   = tid & 31;
  const int laneM  = lane & 15;
  const int laneHi = lane >> 4;   // 0: lanes 0-15, 1: lanes 16-31
  const int mloc   = wave * 16;

  // A fragments: loaded once, reused by every output tile.
  v16bf afrag[KSTEP];
  {
    const bf16_t* xrow = &Xs[mloc + laneM][laneHi * 8];
#pragma unroll
    for (int i = 0; i < KSTEP; ++i) {
      v8bf a0 = *(const v8bf*)(xrow + i * 32);
      v8bf a1 = *(const v8bf*)(xrow + i * 32 + 16);
      afrag[i] = __builtin_shufflevector(a0, a1, 0, 1, 2, 3, 4, 5, 6, 7,
                                         8, 9, 10, 11, 12, 13, 14, 15);
    }
  }

  // per-tile compute: cluster all B loads, then run the WMMA chain
  auto compute_tile = [&](int o0) -> v8f {
    const bf16_t* wrowp = &Ws[o0 + laneM][laneHi * 8];
    v16bf bfrag[KSTEP];
#pragma unroll
    for (int i = 0; i < KSTEP; ++i) {
      v8bf b0 = *(const v8bf*)(wrowp + i * 32);
      v8bf b1 = *(const v8bf*)(wrowp + i * 32 + 16);
      bfrag[i] = __builtin_shufflevector(b0, b1, 0, 1, 2, 3, 4, 5, 6, 7,
                                         8, 9, 10, 11, 12, 13, 14, 15);
    }
    v8f acc = {0.f, 0.f, 0.f, 0.f, 0.f, 0.f, 0.f, 0.f};
#pragma unroll
    for (int i = 0; i < KSTEP; ++i)
      acc = __builtin_amdgcn_wmma_f32_16x16x32_bf16(
          false, afrag[i], false, bfrag[i], (short)0, acc, false, false);
    return acc;
  };

  const int r0 = rowBase + mloc + laneHi * 8;   // C/D: VGPR r -> M=r / M=8+r

  if (rowBase + 128 <= N) {
    // fast path: whole block in-bounds, immediate-offset stores
#pragma unroll
    for (int o0 = 0; o0 < KOUTPAD; o0 += 16) {
      v8f acc = compute_tile(o0);
      const int wrow = o0 + laneM;
      if (KOUT == KOUTPAD || wrow < KOUT) {
        const float bv = bias[wrow];
        float* cbase = C + (size_t)r0 * KOUT + wrow;
#pragma unroll
        for (int r = 0; r < 8; ++r) cbase[r * KOUT] = acc[r] + bv;
      }
    }
  } else {
    // tail block: per-row guard
#pragma unroll
    for (int o0 = 0; o0 < KOUTPAD; o0 += 16) {
      v8f acc = compute_tile(o0);
      const int wrow = o0 + laneM;
      if (KOUT == KOUTPAD || wrow < KOUT) {
        const float bv = bias[wrow];
        float* cbase = C + (size_t)r0 * KOUT + wrow;
#pragma unroll
        for (int r = 0; r < 8; ++r)
          if (r0 + r < N) cbase[r * KOUT] = acc[r] + bv;
      }
    }
  }
}

// ---------------------------------------------------- per-node attention dot
__global__ void k_alpha(const float* __restrict__ xt,
                        const float* __restrict__ attL,
                        const float* __restrict__ attR,
                        float* __restrict__ aL, float* __restrict__ aR, int N) {
  int i = blockIdx.x * blockDim.x + threadIdx.x;
  if (i >= N * HEADS) return;
  int n = i >> 1, h = i & 1;
  const float* row = xt + (size_t)n * FDIM + h * HID;
  float sl = 0.f, sr = 0.f;
#pragma unroll 4
  for (int c = 0; c < HID; ++c) {
    float v = row[c];
    sl += v * attL[c * HEADS + h];
    sr += v * attR[c * HEADS + h];
  }
  aL[i] = sl;
  aR[i] = sr;
}

// ---------------------------------------------------------------- edge passes
__global__ void k_edge_score(const int* __restrict__ ei,
                             const float* __restrict__ aL,
                             const float* __restrict__ aR,
                             float* __restrict__ escr,
                             unsigned* __restrict__ segmax, int E) {
  int e = blockIdx.x * blockDim.x + threadIdx.x;
  if (e >= E) return;
  int s = ei[e], d = ei[E + e];
#pragma unroll
  for (int h = 0; h < HEADS; ++h) {
    float v = aR[d * HEADS + h] + aL[s * HEADS + h];
    v = v > 0.f ? v : SLOPE * v;               // leaky relu
    escr[(size_t)e * HEADS + h] = v;
    atomicMax(&segmax[d * HEADS + h], ford(v));
  }
}

__global__ void k_edge_exp(const int* __restrict__ ei,
                           float* __restrict__ escr,
                           const unsigned* __restrict__ segmax,
                           float* __restrict__ denom, int E) {
  int e = blockIdx.x * blockDim.x + threadIdx.x;
  if (e >= E) return;
  int d = ei[E + e];
#pragma unroll
  for (int h = 0; h < HEADS; ++h) {
    float m  = funord(segmax[d * HEADS + h]);
    float ex = __expf(escr[(size_t)e * HEADS + h] - m);
    escr[(size_t)e * HEADS + h] = ex;
    atomicAdd(&denom[d * HEADS + h], ex);
  }
}

// one thread per (edge, 4 features): float4 gather + 4 L2 atomics
__global__ void k_edge_scatter(const int* __restrict__ ei,
                               const float* __restrict__ escr,
                               const float* __restrict__ denom,
                               const float* __restrict__ xt,
                               float* __restrict__ agg, int E) {
  long long t = (long long)blockIdx.x * blockDim.x + threadIdx.x;
  if (t >= (long long)E * 32) return;
  int e = (int)(t >> 5);
  int c = (int)(t & 31) * 4;
  int h = c >> 6;
  int s = ei[e], d = ei[E + e];
  float al = escr[(size_t)e * HEADS + h] / (denom[d * HEADS + h] + EPSV);
  const float4 xv = *(const float4*)(xt + (size_t)s * FDIM + c);
  float* o = agg + (size_t)d * FDIM + c;
  atomicAdd(o + 0, al * xv.x);
  atomicAdd(o + 1, al * xv.y);
  atomicAdd(o + 2, al * xv.z);
  atomicAdd(o + 3, al * xv.w);
}

// ------------------------------------------------------------- log_softmax
__global__ void k_logsoftmax(const float* __restrict__ logits,
                             float* __restrict__ out, int N) {
  int n = blockIdx.x * blockDim.x + threadIdx.x;
  if (n >= N) return;
  const float* r = logits + (size_t)n * OUTC;
  float m = -3.4e38f;
  for (int c = 0; c < OUTC; ++c) m = fmaxf(m, r[c]);
  float s = 0.f;
  for (int c = 0; c < OUTC; ++c) s += __expf(r[c] - m);
  float ls = __logf(s) + m;
  for (int c = 0; c < OUTC; ++c) out[(size_t)n * OUTC + c] = r[c] - ls;
}

// ================================================================= launcher
extern "C" void kernel_launch(void* const* d_in, const int* in_sizes, int n_in,
                              void* d_out, int out_size, void* d_ws, size_t ws_size,
                              hipStream_t stream) {
  const float* x   = (const float*)d_in[0];
  const int*   ei  = (const int*)d_in[1];
  const float* W1  = (const float*)d_in[2];
  const float* b1  = (const float*)d_in[3];
  const float* al1 = (const float*)d_in[4];
  const float* ar1 = (const float*)d_in[5];
  const float* W2  = (const float*)d_in[6];
  const float* b2  = (const float*)d_in[7];
  const float* al2 = (const float*)d_in[8];
  const float* ar2 = (const float*)d_in[9];
  const float* Wp1 = (const float*)d_in[10];
  const float* bp1 = (const float*)d_in[11];
  const float* Wp2 = (const float*)d_in[12];
  const float* bp2 = (const float*)d_in[13];
  float* out = (float*)d_out;

  const int N = in_sizes[0] / FDIM;
  const int E = in_sizes[1] / 2;

  // workspace carve-out (256B aligned)
  char* w = (char*)d_ws;
  auto alloc = [&](size_t bytes) -> void* {
    void* p = (void*)w;
    w += (bytes + 255) & ~(size_t)255;
    return p;
  };
  bf16_t* Wb1   = (bf16_t*)alloc((size_t)FDIM * FDIM * 2);   // 128x128
  bf16_t* Wb2   = (bf16_t*)alloc((size_t)FDIM * FDIM * 2);   // 128x128
  bf16_t* Wpb1  = (bf16_t*)alloc((size_t)HID * FDIM * 2);    // 64x128
  bf16_t* Wpb2  = (bf16_t*)alloc((size_t)48 * HID * 2);      // 48x64 (40 padded)
  bf16_t* xb    = (bf16_t*)alloc((size_t)N * FDIM * 2);      // bf16 GEMM input
  float*  xt    = (float*)alloc((size_t)N * FDIM * 4);       // GEMM output / h64
  float*  agg   = (float*)alloc((size_t)N * FDIM * 4);       // scatter target
  float*  aL    = (float*)alloc((size_t)N * HEADS * 4);
  float*  aR    = (float*)alloc((size_t)N * HEADS * 4);
  unsigned* smx = (unsigned*)alloc((size_t)N * HEADS * 4);
  float*  den   = (float*)alloc((size_t)N * HEADS * 4);
  float*  escr  = (float*)alloc((size_t)E * HEADS * 4);      // reused as hb64
  bf16_t* hb64  = (bf16_t*)escr;                             // alias (6.4MB each)
  float*  logits = (float*)xb;                               // alias (xb free by then)

  const int B = 256;
  auto cdiv = [](long long a, long long b) { return (unsigned)((a + b - 1) / b); };
  const unsigned gNF   = cdiv((long long)N * FDIM, B);
  const unsigned gNH   = cdiv((long long)N * HEADS, B);
  const unsigned gE    = cdiv(E, B);
  const unsigned gE32  = cdiv((long long)E * 32, B);
  const unsigned gRows = cdiv(N, 128);

  // weight conversion (fp32 -> bf16, padded rows), L2-resident
  k_cvt_pad_bf16<<<cdiv(FDIM * FDIM, B), B, 0, stream>>>(W1, Wb1, FDIM, FDIM, FDIM);
  k_cvt_pad_bf16<<<cdiv(FDIM * FDIM, B), B, 0, stream>>>(W2, Wb2, FDIM, FDIM, FDIM);
  k_cvt_pad_bf16<<<cdiv(HID * FDIM, B), B, 0, stream>>>(Wp1, Wpb1, HID, FDIM, HID);
  k_cvt_pad_bf16<<<cdiv(48 * HID, B), B, 0, stream>>>(Wp2, Wpb2, OUTC, HID, 48);

  // ---------------- GAT layer 1 ----------------
  k_cvt_f32_bf16<<<gNF, B, 0, stream>>>(x, xb, N * FDIM, 0);
  k_gemm_wmma<FDIM, FDIM, FDIM><<<gRows, B, 0, stream>>>(xb, Wb1, b1, xt, N);
  k_alpha<<<gNH, B, 0, stream>>>(xt, al1, ar1, aL, aR, N);
  k_fill_f32<<<gNH, B, 0, stream>>>((float*)smx, 0.0f, N * HEADS);  // 0 == -inf in ord space
  k_fill_f32<<<gNH, B, 0, stream>>>(den, 0.0f, N * HEADS);
  k_fill_f32<<<gNF, B, 0, stream>>>(agg, 0.0f, N * FDIM);
  k_edge_score<<<gE, B, 0, stream>>>(ei, aL, aR, escr, smx, E);
  k_edge_exp<<<gE, B, 0, stream>>>(ei, escr, smx, den, E);
  k_edge_scatter<<<gE32, B, 0, stream>>>(ei, escr, den, xt, agg, E);

  // ---------------- GAT layer 2 ----------------
  k_cvt_f32_bf16<<<gNF, B, 0, stream>>>(agg, xb, N * FDIM, 1);      // relu fused
  k_gemm_wmma<FDIM, FDIM, FDIM><<<gRows, B, 0, stream>>>(xb, Wb2, b2, xt, N);
  k_alpha<<<gNH, B, 0, stream>>>(xt, al2, ar2, aL, aR, N);
  k_fill_f32<<<gNH, B, 0, stream>>>((float*)smx, 0.0f, N * HEADS);
  k_fill_f32<<<gNH, B, 0, stream>>>(den, 0.0f, N * HEADS);
  k_fill_f32<<<gNF, B, 0, stream>>>(agg, 0.0f, N * FDIM);
  k_edge_score<<<gE, B, 0, stream>>>(ei, aL, aR, escr, smx, E);
  k_edge_exp<<<gE, B, 0, stream>>>(ei, escr, smx, den, E);
  k_edge_scatter<<<gE32, B, 0, stream>>>(ei, escr, den, xt, agg, E);

  // ---------------- post-MLP + log_softmax ----------------
  k_cvt_f32_bf16<<<gNF, B, 0, stream>>>(agg, xb, N * FDIM, 1);      // relu fused
  k_gemm_wmma<FDIM, HID, HID><<<gRows, B, 0, stream>>>(xb, Wpb1, bp1, xt, N);
  k_cvt_f32_bf16<<<cdiv((long long)N * HID, B), B, 0, stream>>>(xt, hb64, N * HID, 0);
  k_gemm_wmma<HID, 48, OUTC><<<gRows, B, 0, stream>>>(hb64, Wpb2, bp2, logits, N);
  k_logsoftmax<<<cdiv(N, B), B, 0, stream>>>(logits, out, N);
}
